// FactoredHmmLm_77567109365802
// MI455X (gfx1250) — matrix operand order repaired and checked by reading, not compiled, not run on hardware
//
#include <hip/hip_runtime.h>

#define V_   10000
#define C_   8192
#define NC_  64
#define SPW_ 128
#define H_   256
#define B_   16
#define T_   33
#define NBT_ (B_*(T_-1))

typedef __bf16 bf16_t;
typedef __attribute__((ext_vector_type(16))) __bf16 v16bf;
typedef __attribute__((ext_vector_type(8)))  __bf16 v8bf;
typedef __attribute__((ext_vector_type(8)))  float  v8f;

// ---- WMMA fragment loaders (CDNA5 ISA 7.12.2 layouts, wave32) ----
// A (16x32 bf16): lane l -> row l%16; K = (h&7) + 16*(h>>3) + 8*(l>>4)
static __device__ __forceinline__ v16bf load_a_frag(const bf16_t* __restrict__ rowp, int kk, int hi) {
  const v8bf lo  = *reinterpret_cast<const v8bf*>(rowp + kk + hi * 8);
  const v8bf hiv = *reinterpret_cast<const v8bf*>(rowp + kk + hi * 8 + 16);
  return __builtin_shufflevector(lo, hiv, 0,1,2,3,4,5,6,7,8,9,10,11,12,13,14,15);
}
// B (32x16 bf16): lane l -> col l%16; K = (h&7) + 16*(h>>3) + 8*(l>>4)
static __device__ __forceinline__ v16bf load_b_frag(const bf16_t* __restrict__ rowp, int kk, int hi) {
  const v8bf lo  = *reinterpret_cast<const v8bf*>(rowp + kk + hi * 8);
  const v8bf hiv = *reinterpret_cast<const v8bf*>(rowp + kk + hi * 8 + 16);
  return __builtin_shufflevector(lo, hiv, 0,1,2,3,4,5,6,7,8,9,10,11,12,13,14,15);
}

static __device__ __forceinline__ void lse_update(float& m, float& s, float v) {
  float nm = fmaxf(m, v);
  s = s * __expf(m - nm) + __expf(v - nm);
  m = nm;
}

// ---- f32 -> bf16 converter ----
__global__ void f32_to_bf16_kernel(const float* __restrict__ in, bf16_t* __restrict__ out, int n) {
  for (int i = blockIdx.x * blockDim.x + threadIdx.x; i < n; i += gridDim.x * blockDim.x)
    out[i] = (bf16_t)in[i];
}

// ---- word -> cluster ----
__global__ void word_cluster_kernel(const int* __restrict__ w2s, int* __restrict__ wcl) {
  int v = blockIdx.x * blockDim.x + threadIdx.x;
  if (v < V_) wcl[v] = w2s[(size_t)v * SPW_] >> 7;
}

// ---- OUT[m,n] = relu(A@Wt^T + bias)[m,n] (+ resid)  -> f32 and/or bf16 ----
__global__ void gemm_bt_kernel(const bf16_t* __restrict__ A, const bf16_t* __restrict__ Wt,
                               const float* __restrict__ bias, const float* __restrict__ resid,
                               float* __restrict__ out32, bf16_t* __restrict__ out16,
                               int M, int N, int K) {
  const int ntn = N >> 4;
  const int mt = blockIdx.x / ntn;
  const int nt = blockIdx.x % ntn;
  const int lane = threadIdx.x, l15 = lane & 15, hi = lane >> 4;
  const bf16_t* arow = A  + (size_t)(mt * 16 + l15) * K;
  const bf16_t* brow = Wt + (size_t)(nt * 16 + l15) * K;
  v8f acc = {};
  for (int kk = 0; kk < K; kk += 32) {
    v16bf a = load_a_frag(arow, kk, hi);
    v16bf b = load_b_frag(brow, kk, hi);
    acc = __builtin_amdgcn_wmma_f32_16x16x32_bf16(false, a, false, b, (short)0, acc, false, false);
  }
  const int colO = nt * 16 + l15;
  const float bv = bias[colO];
#pragma unroll
  for (int r = 0; r < 8; ++r) {
    const int rowO = mt * 16 + r + 8 * hi;
    float v = acc[r] + bv;
    v = v > 0.f ? v : 0.f;
    if (resid) v += resid[(size_t)rowO * N + colO];
    if (out32) out32[(size_t)rowO * N + colO] = v;
    if (out16) out16[(size_t)rowO * N + colO] = (bf16_t)v;
  }
}

// ---- start logits: slog[i] = r32[i,:] . sow + sob ----
__global__ void start_logit_kernel(const float* __restrict__ r32, const float* __restrict__ sow,
                                   const float* __restrict__ sob, float* __restrict__ slog) {
  int i = blockIdx.x * blockDim.x + threadIdx.x;
  if (i >= C_) return;
  const float* row = r32 + (size_t)i * H_;
  float acc = sob[0];
  for (int k = 0; k < H_; k += 4) {
    float4 a = *reinterpret_cast<const float4*>(row + k);
    float4 w = *reinterpret_cast<const float4*>(sow + k);
    acc += a.x * w.x + a.y * w.y + a.z * w.z + a.w * w.w;
  }
  slog[i] = acc;
}

// ---- single-block logsumexp over n values ----
__global__ void lse_all_kernel(const float* __restrict__ x, int n, float* __restrict__ out) {
  __shared__ float sm[256], ss[256];
  int t = threadIdx.x;
  float m = -__builtin_inff(), s = 0.f;
  for (int i = t; i < n; i += 256) lse_update(m, s, x[i]);
  sm[t] = m; ss[t] = s; __syncthreads();
  for (int st = 128; st > 0; st >>= 1) {
    if (t < st) {
      float om = sm[t + st], os = ss[t + st];
      float nm = fmaxf(sm[t], om);
      ss[t] = ss[t] * __expf(sm[t] - nm) + os * __expf(om - nm);
      sm[t] = nm;
    }
    __syncthreads();
  }
  if (t == 0) out[0] = sm[0] + __logf(ss[0]);
}

// ---- transition denominators: streamed WMMA GEMM + online LSE, no 256MB matrix ----
// grid = (C/16 row tiles) x 16 column chunks (512 cols each); block = 1 wave.
__global__ void trans_lse_partial_kernel(const bf16_t* __restrict__ Ares,
                                         const bf16_t* __restrict__ Nse,
                                         float2* __restrict__ part) {
  const int rt = blockIdx.x >> 4;
  const int jc = blockIdx.x & 15;
  const int lane = threadIdx.x, l15 = lane & 15, hi = lane >> 4;
  const bf16_t* arow = Ares + (size_t)(rt * 16 + l15) * H_;
  v16bf afr[8];
#pragma unroll
  for (int k8 = 0; k8 < 8; ++k8) afr[k8] = load_a_frag(arow, k8 * 32, hi);
  float m[8], s[8];
#pragma unroll
  for (int r = 0; r < 8; ++r) { m[r] = -__builtin_inff(); s[r] = 0.f; }
  for (int jt = 0; jt < 32; ++jt) {
    const bf16_t* brow = Nse + (size_t)(jc * 512 + jt * 16 + l15) * H_;
    v8f c = {};
#pragma unroll
    for (int k8 = 0; k8 < 8; ++k8) {
      v16bf b = load_b_frag(brow, k8 * 32, hi);
      c = __builtin_amdgcn_wmma_f32_16x16x32_bf16(false, afr[k8], false, b, (short)0, c, false, false);
    }
#pragma unroll
    for (int r = 0; r < 8; ++r) lse_update(m[r], s[r], c[r]);
  }
  // cross-lane LSE merge within each 16-lane column group (masks 1,2,4,8)
#pragma unroll
  for (int r = 0; r < 8; ++r) {
#pragma unroll
    for (int mask = 1; mask <= 8; mask <<= 1) {
      float om = __shfl_xor(m[r], mask, 32);
      float os = __shfl_xor(s[r], mask, 32);
      float nm = fmaxf(m[r], om);
      s[r] = s[r] * __expf(m[r] - nm) + os * __expf(om - nm);
      m[r] = nm;
    }
  }
  if (l15 == 0) {
#pragma unroll
    for (int r = 0; r < 8; ++r) {
      int row = rt * 16 + r + 8 * hi;
      part[(size_t)row * 16 + jc] = make_float2(m[r], s[r]);
    }
  }
}

__global__ void lse_merge16_kernel(const float2* __restrict__ part, float* __restrict__ out, int n) {
  int i = blockIdx.x * blockDim.x + threadIdx.x;
  if (i >= n) return;
  float m = -__builtin_inff(), s = 0.f;
  for (int c = 0; c < 16; ++c) {
    float2 p = part[(size_t)i * 16 + c];
    float nm = fmaxf(m, p.x);
    s = s * __expf(m - nm) + p.y * __expf(p.x - nm);
    m = nm;
  }
  out[i] = m + __logf(s);
}

// ---- emission: valid logits only (word v x its 128 cluster states) ----
__global__ void emis_logit_kernel(const float* __restrict__ eres, const float* __restrict__ pw,
                                  const float* __restrict__ pb, const int* __restrict__ wcl,
                                  float* __restrict__ elog) {
  const int v = blockIdx.x, lane = threadIdx.x;
  const int cl = wcl[v];
  const float* wrow = pw + (size_t)v * H_;
#pragma unroll
  for (int q = 0; q < 4; ++q) {
    int st = lane + q * 32;
    const float* erow = eres + (size_t)(cl * SPW_ + st) * H_;
    float acc = 0.f;
    for (int k = 0; k < H_; k += 4) {
      float4 a = *reinterpret_cast<const float4*>(erow + k);
      float4 w = *reinterpret_cast<const float4*>(wrow + k);
      acc += a.x * w.x + a.y * w.y + a.z * w.z + a.w * w.w;
    }
    elog[(size_t)v * SPW_ + st] = acc + pb[v];
  }
}

// ---- per-state emission denominator: LSE over cluster's words (exp(NEG)==0 exactly) ----
__global__ void emis_denom_kernel(const float* __restrict__ elog, const int* __restrict__ wcl,
                                  float* __restrict__ eden) {
  const int c = blockIdx.x;
  const int cl = c >> 7, s = c & 127;
  const int lane = threadIdx.x;
  float m = -__builtin_inff(), sum = 0.f;
  for (int v = lane; v < V_; v += 32)
    if (wcl[v] == cl) lse_update(m, sum, elog[(size_t)v * SPW_ + s]);
#pragma unroll
  for (int mask = 1; mask <= 16; mask <<= 1) {
    float om = __shfl_xor(m, mask, 32);
    float os = __shfl_xor(sum, mask, 32);
    float nm = fmaxf(m, om);
    float e1 = (m  == -__builtin_inff()) ? 0.f : __expf(m  - nm);
    float e2 = (om == -__builtin_inff()) ? 0.f : __expf(om - nm);
    sum = sum * e1 + os * e2;
    m = nm;
  }
  if (lane == 0) eden[c] = (sum > 0.f) ? (m + __logf(sum)) : -1e30f;
}

// ---- obs[b,t,s] = elog[text, s] - eden[state] ----
__global__ void obs_kernel(const int* __restrict__ text, const int* __restrict__ wcl,
                           const float* __restrict__ elog, const float* __restrict__ eden,
                           float* __restrict__ obs) {
  int g = blockIdx.x * blockDim.x + threadIdx.x;
  if (g >= B_ * T_ * SPW_) return;
  int tok = g >> 7, s = g & 127;
  int w = text[tok];
  obs[g] = elog[(size_t)w * SPW_ + s] - eden[wcl[w] * SPW_ + s];
}

// ---- transition blocks for each (b,t): 128x128 WMMA GEMM, stored transposed (lpT[bt][j][i]) ----
__global__ void lp_block_kernel(const bf16_t* __restrict__ tres, const bf16_t* __restrict__ nse,
                                const float* __restrict__ tden, const int* __restrict__ text,
                                const int* __restrict__ wcl, float* __restrict__ lpT) {
  const int bid = blockIdx.x;
  const int bt = bid >> 6, tile = bid & 63;
  const int ti = tile >> 3, tj = tile & 7;
  const int b = bt / (T_ - 1), t = bt % (T_ - 1);
  const int clp = wcl[text[b * T_ + t]];
  const int cln = wcl[text[b * T_ + t + 1]];
  const int lane = threadIdx.x, l15 = lane & 15, hi = lane >> 4;
  const bf16_t* arow = tres + (size_t)(clp * SPW_ + ti * 16 + l15) * H_;
  const bf16_t* brow = nse  + (size_t)(cln * SPW_ + tj * 16 + l15) * H_;
  v8f c = {};
  for (int kk = 0; kk < H_; kk += 32) {
    v16bf a = load_a_frag(arow, kk, hi);
    v16bf bb = load_b_frag(brow, kk, hi);
    c = __builtin_amdgcn_wmma_f32_16x16x32_bf16(false, a, false, bb, (short)0, c, false, false);
  }
  const int jg = tj * 16 + l15;
  float* outrow = lpT + (size_t)bt * (SPW_ * SPW_) + (size_t)jg * SPW_ + ti * 16 + 8 * hi;
  const float* td = tden + clp * SPW_ + ti * 16 + 8 * hi;
#pragma unroll
  for (int r = 0; r < 8; ++r) outrow[r] = c[r] - td[r];
}

// ---- HMM forward scan per batch: alpha_t[j] = LSE_i(lpT[t][j][i] + alpha[i]) + obs[t+1][j] ----
__global__ void forward_kernel(const float* __restrict__ lpT, const float* __restrict__ obs,
                               const float* __restrict__ slog, const float* __restrict__ sden,
                               const int* __restrict__ text, const int* __restrict__ wcl,
                               float* __restrict__ out) {
  __shared__ float alpha[SPW_];
  const int b = blockIdx.x, j = threadIdx.x;
  const int cl0 = wcl[text[b * T_]];
  const float sd = sden[0];
  {
    const float* row = lpT + (size_t)(b * (T_ - 1)) * (SPW_ * SPW_) + (size_t)j * SPW_;
    const float* o0  = obs + (size_t)(b * T_) * SPW_;
    float m = -__builtin_inff(), s = 0.f;
    for (int i = 0; i < SPW_; ++i)
      lse_update(m, s, row[i] + (slog[cl0 * SPW_ + i] - sd) + o0[i]);
    alpha[j] = m + __logf(s) + obs[(size_t)(b * T_ + 1) * SPW_ + j];
  }
  __syncthreads();
  for (int t = 1; t < T_ - 1; ++t) {
    const float* row = lpT + (size_t)(b * (T_ - 1) + t) * (SPW_ * SPW_) + (size_t)j * SPW_;
    float m = -__builtin_inff(), s = 0.f;
    for (int i = 0; i < SPW_; ++i) lse_update(m, s, row[i] + alpha[i]);
    float a = m + __logf(s) + obs[(size_t)(b * T_ + t + 1) * SPW_ + j];
    __syncthreads();
    alpha[j] = a;
    __syncthreads();
  }
  if (j == 0) {
    float m = -__builtin_inff(), s = 0.f;
    for (int i = 0; i < SPW_; ++i) lse_update(m, s, alpha[i]);
    out[b] = m + __logf(s);
  }
}

extern "C" void kernel_launch(void* const* d_in, const int* in_sizes, int n_in,
                              void* d_out, int out_size, void* d_ws, size_t ws_size,
                              hipStream_t stream) {
  (void)in_sizes; (void)n_in; (void)out_size; (void)ws_size;
  const int*   text      = (const int*)d_in[0];
  const int*   w2s       = (const int*)d_in[1];
  const float* start_emb = (const float*)d_in[2];
  const float* sw1 = (const float*)d_in[3];
  const float* sb1 = (const float*)d_in[4];
  const float* sw2 = (const float*)d_in[5];
  const float* sb2 = (const float*)d_in[6];
  const float* sow = (const float*)d_in[7];
  const float* sob = (const float*)d_in[8];
  const float* state_emb = (const float*)d_in[9];
  const float* tw1 = (const float*)d_in[10];
  const float* tb1 = (const float*)d_in[11];
  const float* tw2 = (const float*)d_in[12];
  const float* tb2 = (const float*)d_in[13];
  const float* nse = (const float*)d_in[14];
  const float* pre = (const float*)d_in[15];
  const float* ew1 = (const float*)d_in[16];
  const float* eb1 = (const float*)d_in[17];
  const float* ew2 = (const float*)d_in[18];
  const float* eb2 = (const float*)d_in[19];
  const float* pw  = (const float*)d_in[20];
  const float* pb  = (const float*)d_in[21];
  float* out = (float*)d_out;

  char* base = (char*)d_ws;
  size_t off = 0;
  auto carve = [&](size_t bytes) -> char* {
    char* p = base + off;
    off = (off + bytes + 255) & ~(size_t)255;
    return p;
  };
  bf16_t* x_bf    = (bf16_t*)carve((size_t)C_ * H_ * 2);
  bf16_t* h1_bf   = (bf16_t*)carve((size_t)C_ * H_ * 2);
  bf16_t* w1_bf   = (bf16_t*)carve((size_t)H_ * H_ * 2);
  bf16_t* w2_bf   = (bf16_t*)carve((size_t)H_ * H_ * 2);
  bf16_t* tres_bf = (bf16_t*)carve((size_t)C_ * H_ * 2);
  bf16_t* nse_bf  = (bf16_t*)carve((size_t)C_ * H_ * 2);
  float*  r32     = (float*)carve((size_t)C_ * H_ * 4);   // rstart32, later eres32 (disjoint lifetimes)
  float*  slog    = (float*)carve((size_t)C_ * 4);
  float*  sden    = (float*)carve(256);
  float2* tpart   = (float2*)carve((size_t)C_ * 16 * 8);
  float*  tden    = (float*)carve((size_t)C_ * 4);
  float*  elog    = (float*)carve((size_t)V_ * SPW_ * 4);
  float*  eden    = (float*)carve((size_t)C_ * 4);
  float*  obsb    = (float*)carve((size_t)B_ * T_ * SPW_ * 4);
  int*    wcl     = (int*)carve((size_t)V_ * 4);
  float*  lpT     = (float*)carve((size_t)NBT_ * SPW_ * SPW_ * 4);

  const int TPB = 256;
  const int GEMM_GRID = (C_ / 16) * (H_ / 16);

  word_cluster_kernel<<<(V_ + TPB - 1) / TPB, TPB, 0, stream>>>(w2s, wcl);

  // -------- start pipeline --------
  f32_to_bf16_kernel<<<2048, TPB, 0, stream>>>(start_emb, x_bf, C_ * H_);
  f32_to_bf16_kernel<<<256, TPB, 0, stream>>>(sw1, w1_bf, H_ * H_);
  f32_to_bf16_kernel<<<256, TPB, 0, stream>>>(sw2, w2_bf, H_ * H_);
  gemm_bt_kernel<<<GEMM_GRID, 32, 0, stream>>>(x_bf, w1_bf, sb1, nullptr, nullptr, h1_bf, C_, H_, H_);
  gemm_bt_kernel<<<GEMM_GRID, 32, 0, stream>>>(h1_bf, w2_bf, sb2, start_emb, r32, nullptr, C_, H_, H_);
  start_logit_kernel<<<C_ / TPB, TPB, 0, stream>>>(r32, sow, sob, slog);
  lse_all_kernel<<<1, 256, 0, stream>>>(slog, C_, sden);

  // -------- transition pipeline --------
  f32_to_bf16_kernel<<<2048, TPB, 0, stream>>>(state_emb, x_bf, C_ * H_);
  f32_to_bf16_kernel<<<256, TPB, 0, stream>>>(tw1, w1_bf, H_ * H_);
  f32_to_bf16_kernel<<<256, TPB, 0, stream>>>(tw2, w2_bf, H_ * H_);
  f32_to_bf16_kernel<<<2048, TPB, 0, stream>>>(nse, nse_bf, C_ * H_);
  gemm_bt_kernel<<<GEMM_GRID, 32, 0, stream>>>(x_bf, w1_bf, tb1, nullptr, nullptr, h1_bf, C_, H_, H_);
  gemm_bt_kernel<<<GEMM_GRID, 32, 0, stream>>>(h1_bf, w2_bf, tb2, state_emb, nullptr, tres_bf, C_, H_, H_);
  trans_lse_partial_kernel<<<(C_ / 16) * 16, 32, 0, stream>>>(tres_bf, nse_bf, tpart);
  lse_merge16_kernel<<<(C_ + TPB - 1) / TPB, TPB, 0, stream>>>(tpart, tden, C_);

  // -------- emission pipeline --------
  f32_to_bf16_kernel<<<2048, TPB, 0, stream>>>(pre, x_bf, C_ * H_);
  f32_to_bf16_kernel<<<256, TPB, 0, stream>>>(ew1, w1_bf, H_ * H_);
  f32_to_bf16_kernel<<<256, TPB, 0, stream>>>(ew2, w2_bf, H_ * H_);
  gemm_bt_kernel<<<GEMM_GRID, 32, 0, stream>>>(x_bf, w1_bf, eb1, nullptr, nullptr, h1_bf, C_, H_, H_);
  gemm_bt_kernel<<<GEMM_GRID, 32, 0, stream>>>(h1_bf, w2_bf, eb2, pre, r32, nullptr, C_, H_, H_);
  emis_logit_kernel<<<V_, 32, 0, stream>>>(r32, pw, pb, wcl, elog);
  emis_denom_kernel<<<C_, 32, 0, stream>>>(elog, wcl, eden);
  obs_kernel<<<(B_ * T_ * SPW_ + TPB - 1) / TPB, TPB, 0, stream>>>(text, wcl, elog, eden, obsb);

  // -------- per-step transition blocks + forward scan --------
  lp_block_kernel<<<NBT_ * 64, 32, 0, stream>>>(tres_bf, nse_bf, tden, text, wcl, lpT);
  forward_kernel<<<B_, SPW_, 0, stream>>>(lpT, obsb, slog, sden, text, wcl, out);
}
